// FourierFT_79559974191643
// MI455X (gfx1250) — matrix operand order repaired and checked by reading, not compiled, not run on hardware
//
#include <hip/hip_runtime.h>
#include <hip/hip_bf16.h>

#define D1V 4096
#define NMODES 100
#define MTOT 8192
#define KDIM 4096
#define NDIM 4096

typedef __attribute__((ext_vector_type(16))) _Float16 v16h;
typedef __attribute__((ext_vector_type(8)))  _Float16 h8;
typedef __attribute__((ext_vector_type(8)))  float    v8f;
typedef __attribute__((ext_vector_type(4)))  float    f4;
typedef __attribute__((ext_vector_type(4)))  int      v4i;

union V16H { v16h v; h8 h2[2]; _Float16 e[16]; };
union AF16 { f4 q[4]; float e[16]; };

// ---- CDNA5 async global->LDS path (guarded; falls back to reg-staged) ----
#if defined(__has_builtin)
#if __has_builtin(__builtin_amdgcn_global_load_async_to_lds_b128)
#define HAVE_ASYNC_LDS 1
#endif
#if __has_builtin(__builtin_amdgcn_s_wait_asynccnt)
#define HAVE_WAIT_ASYNC 1
#endif
#endif

#if defined(HAVE_ASYNC_LDS)
// signature learned from clang diagnostic: param 0 is `int4 __device__*` (AS1)
typedef __attribute__((address_space(1))) v4i* gv4p;
typedef __attribute__((address_space(3))) v4i* lv4p;
__device__ __forceinline__ void async_cp16(const void* g, void* l) {
  __builtin_amdgcn_global_load_async_to_lds_b128((gv4p)g, (lv4p)l, 0, 0);
}
__device__ __forceinline__ void wait_async0() {
#if defined(HAVE_WAIT_ASYNC)
  __builtin_amdgcn_s_wait_asynccnt(0);
#else
  asm volatile("s_wait_asynccnt 0x0" ::: "memory");
#endif
}
#endif

// ---------------------------------------------------------------------------
// Kernel 1: rank-200 trig factors of the sparse inverse FFT.
//   Delta_W[k][l] = dot(U[k], V[l])  via cos(A+B) = cosA cosB - sinA sinB,
//   with ALPHA/(D1*D2) and c_j folded into U. Exact integer phase mod 4096.
// ---------------------------------------------------------------------------
__global__ void fourier_trig_kernel(const int* __restrict__ E,
                                    const float* __restrict__ c,
                                    float* __restrict__ U,
                                    float* __restrict__ V) {
  int idx = blockIdx.x * blockDim.x + threadIdx.x;   // D1*NMODES threads
  if (idx >= D1V * NMODES) return;
  int j = idx % NMODES;
  int k = idx / NMODES;
  const float w = 6.28318530717958647692f / 4096.0f;
  const float scale = 300.0f / (4096.0f * 4096.0f);
  int e0 = E[j];
  int e1 = E[NMODES + j];
  float cj = c[j] * scale;
  int p0 = (k * e0) & 4095;
  float a0 = (float)p0 * w;
  U[k * (2 * NMODES) + 2 * j]     =  cj * __cosf(a0);
  U[k * (2 * NMODES) + 2 * j + 1] = -cj * __sinf(a0);
  int p1 = (k * e1) & 4095;
  float a1 = (float)p1 * w;
  V[k * (2 * NMODES) + 2 * j]     = __cosf(a1);
  V[k * (2 * NMODES) + 2 * j + 1] = __sinf(a1);
}

// ---------------------------------------------------------------------------
// Kernel 2: Wt[n][k] = W_base[n][k] + Delta_W[k][n], split into f16 hi/lo.
// ---------------------------------------------------------------------------
__global__ void fourier_weight_kernel(const float* __restrict__ Wb,
                                      const float* __restrict__ U,
                                      const float* __restrict__ V,
                                      _Float16* __restrict__ Whi,
                                      _Float16* __restrict__ Wlo) {
  int k = blockIdx.x * 256 + threadIdx.x;
  int n0 = blockIdx.y * 32;
  const float* urow = &U[(size_t)k * 200];
  for (int nn = 0; nn < 32; ++nn) {
    int n = n0 + nn;
    const float* vrow = &V[(size_t)n * 200];
    float acc = Wb[(size_t)n * KDIM + k];
#pragma unroll 8
    for (int j = 0; j < 200; ++j) acc = fmaf(urow[j], vrow[j], acc);
    _Float16 hi = (_Float16)acc;
    Whi[(size_t)n * KDIM + k] = hi;
    Wlo[(size_t)n * KDIM + k] = (_Float16)(acc - (float)hi);
  }
}

// ---------------------------------------------------------------------------
// Kernel 3: Out[m][n] = sum_k X[m][k]*Wt[n][k] + bias[n]
// 3-term split-f16 WMMA, 128x128 block / BK=32, 8 waves, 2x4 frags per wave.
// Double-buffered LDS; fill uses GLOBAL_LOAD_ASYNC_TO_LDS_B128 (ASYNCcnt)
// when available, else global->VGPR->LDS staging.
// ---------------------------------------------------------------------------
__launch_bounds__(256)
__global__ void fourier_gemm_kernel(const float* __restrict__ X,
                                    const _Float16* __restrict__ Whi,
                                    const _Float16* __restrict__ Wlo,
                                    const float* __restrict__ bias,
                                    float* __restrict__ Out) {
  __shared__ float    As[2][128][36];   // +4 f32 pad: bank stagger
  __shared__ _Float16 Bh[2][128][40];   // +8 half pad
  __shared__ _Float16 Bl[2][128][40];

  const int tid  = threadIdx.x;
  const int lane = tid & 31;
  const int wid  = tid >> 5;
  const int lrow = lane & 15;
  const int lhi  = lane >> 4;

  const int blockN = blockIdx.x * 128;
  const int blockM = blockIdx.y * 128;
  const int waveM  = (wid & 3) * 32;
  const int waveN  = (wid >> 2) * 64;

  // per-thread copy mapping (same for both fill paths)
  const int arow0 = tid >> 3, akq0 = tid & 7;        // A: 4 chunks, +32 rows apart
  const int brow0 = tid >> 2, bcq0 = tid & 3;        // B: 2 chunks, +64 rows apart

  v8f acc[2][4];
#pragma unroll
  for (int i = 0; i < 2; ++i)
#pragma unroll
    for (int j = 0; j < 4; ++j)
#pragma unroll
      for (int r = 0; r < 8; ++r) acc[i][j][r] = 0.0f;

#if defined(HAVE_ASYNC_LDS)
  // ---- async fill: global -> LDS directly, tracked by ASYNCcnt ----
  auto fill_tile = [&](int buf, int kt) {
#pragma unroll
    for (int p = 0; p < 4; ++p) {
      int arow = arow0 + p * 32;
      async_cp16(&X[(size_t)(blockM + arow) * KDIM + kt + akq0 * 4],
                 &As[buf][arow][akq0 * 4]);
    }
#pragma unroll
    for (int p = 0; p < 2; ++p) {
      int brow = brow0 + p * 64;
      size_t goff = (size_t)(blockN + brow) * KDIM + kt + bcq0 * 8;
      async_cp16(&Whi[goff], &Bh[buf][brow][bcq0 * 8]);
      async_cp16(&Wlo[goff], &Bl[buf][brow][bcq0 * 8]);
    }
  };
  fill_tile(0, 0);
  wait_async0();
  __syncthreads();
#else
  // ---- fallback prologue: reg-staged fill of buffer 0 ----
#pragma unroll
  for (int p = 0; p < 4; ++p) {
    int arow = arow0 + p * 32;
    *(f4*)&As[0][arow][akq0 * 4] =
        *(const f4*)&X[(size_t)(blockM + arow) * KDIM + akq0 * 4];
  }
#pragma unroll
  for (int p = 0; p < 2; ++p) {
    int brow = brow0 + p * 64;
    size_t goff = (size_t)(blockN + brow) * KDIM + bcq0 * 8;
    *(h8*)&Bh[0][brow][bcq0 * 8] = *(const h8*)&Whi[goff];
    *(h8*)&Bl[0][brow][bcq0 * 8] = *(const h8*)&Wlo[goff];
  }
  __syncthreads();
#endif

  const int KT = KDIM / 32;  // 128 K-steps
#if !defined(HAVE_ASYNC_LDS)
  f4 pa[4];
  h8 pbh[2], pbl[2];
#endif

  for (int t = 0; t < KT; ++t) {
    const int cur = t & 1, nxt = cur ^ 1;
    const bool havnext = (t + 1) < KT;

    if (havnext) {
      const int kt = (t + 1) * 32;
#if defined(HAVE_ASYNC_LDS)
      fill_tile(nxt, kt);            // fire-and-forget, hidden under WMMAs
#else
#pragma unroll
      for (int p = 0; p < 4; ++p) {
        int arow = arow0 + p * 32;
        pa[p] = *(const f4*)&X[(size_t)(blockM + arow) * KDIM + kt + akq0 * 4];
      }
#pragma unroll
      for (int p = 0; p < 2; ++p) {
        int brow = brow0 + p * 64;
        size_t goff = (size_t)(blockN + brow) * KDIM + kt + bcq0 * 8;
        pbh[p] = *(const h8*)&Whi[goff];
        pbl[p] = *(const h8*)&Wlo[goff];
      }
#endif
      if (t + 2 < KT) {              // speculative L2 prefetch two tiles ahead
        __builtin_prefetch(&X[(size_t)(blockM + arow0) * KDIM + (t + 2) * 32], 0, 1);
        __builtin_prefetch(&Whi[(size_t)(blockN + brow0) * KDIM + (t + 2) * 32], 0, 1);
      }
    }

    // ---- A fragments: 16 f32/lane -> hi/lo f16 per ISA A layout ----
    V16H ahi[2], alo[2];
#pragma unroll
    for (int i = 0; i < 2; ++i) {
      const float* ap = &As[cur][waveM + 16 * i + lrow][lhi * 8];
      AF16 af;
      af.q[0] = *(const f4*)(ap);
      af.q[1] = *(const f4*)(ap + 4);
      af.q[2] = *(const f4*)(ap + 16);
      af.q[3] = *(const f4*)(ap + 20);
#pragma unroll
      for (int q = 0; q < 16; ++q) {
        float f = af.e[q];
        _Float16 h = (_Float16)f;
        ahi[i].e[q] = h;
        alo[i].e[q] = (_Float16)(f - (float)h);
      }
    }

    // ---- B fragments + 24 WMMAs per K-step ----
#pragma unroll
    for (int j = 0; j < 4; ++j) {
      V16H bh, bl;
      const _Float16* bhp = &Bh[cur][waveN + 16 * j + lrow][lhi * 8];
      const _Float16* blp = &Bl[cur][waveN + 16 * j + lrow][lhi * 8];
      bh.h2[0] = *(const h8*)bhp;  bh.h2[1] = *(const h8*)(bhp + 16);
      bl.h2[0] = *(const h8*)blp;  bl.h2[1] = *(const h8*)(blp + 16);
#pragma unroll
      for (int i = 0; i < 2; ++i) {
        acc[i][j] = __builtin_amdgcn_wmma_f32_16x16x32_f16(
            false, ahi[i].v, false, bh.v, (short)0, acc[i][j], false, false);
        acc[i][j] = __builtin_amdgcn_wmma_f32_16x16x32_f16(
            false, ahi[i].v, false, bl.v, (short)0, acc[i][j], false, false);
        acc[i][j] = __builtin_amdgcn_wmma_f32_16x16x32_f16(
            false, alo[i].v, false, bh.v, (short)0, acc[i][j], false, false);
      }
    }

#if defined(HAVE_ASYNC_LDS)
    wait_async0();                   // this wave's async writes to `nxt` done
#else
    if (havnext) {                   // drain prefetch regs into other buffer
#pragma unroll
      for (int p = 0; p < 4; ++p) {
        int arow = arow0 + p * 32;
        *(f4*)&As[nxt][arow][akq0 * 4] = pa[p];
      }
#pragma unroll
      for (int p = 0; p < 2; ++p) {
        int brow = brow0 + p * 64;
        *(h8*)&Bh[nxt][brow][bcq0 * 8] = pbh[p];
        *(h8*)&Bl[nxt][brow][bcq0 * 8] = pbl[p];
      }
    }
#endif
    __syncthreads();
  }

  // ---- epilogue: C/D layout lane=(col), VGPR r -> row r + 8*(lane>>4) ----
#pragma unroll
  for (int j = 0; j < 4; ++j) {
    int col = blockN + waveN + 16 * j + lrow;
    float bv = bias[col];
#pragma unroll
    for (int i = 0; i < 2; ++i) {
      int mbase = blockM + waveM + 16 * i + 8 * lhi;
#pragma unroll
      for (int r = 0; r < 8; ++r)
        Out[(size_t)(mbase + r) * NDIM + col] = acc[i][j][r] + bv;
    }
  }
}

// ---------------------------------------------------------------------------
extern "C" void kernel_launch(void* const* d_in, const int* in_sizes, int n_in,
                              void* d_out, int out_size, void* d_ws, size_t ws_size,
                              hipStream_t stream) {
  (void)in_sizes; (void)n_in; (void)out_size; (void)ws_size;
  const float* x  = (const float*)d_in[0];   // [4,2048,4096]
  const float* c  = (const float*)d_in[1];   // [100]
  const int*   E  = (const int*)d_in[2];     // [2,100]
  const float* Wb = (const float*)d_in[3];   // [4096,4096] (out,in)
  const float* bb = (const float*)d_in[4];   // [4096]
  float* out = (float*)d_out;                // [4,2048,4096]

  // workspace layout (f32 U,V then f16 hi/lo weights): ~74 MB
  float* U = (float*)d_ws;                               // 4096*200
  float* V = U + (size_t)D1V * 2 * NMODES;               // 4096*200
  _Float16* Whi = (_Float16*)(V + (size_t)D1V * 2 * NMODES);
  _Float16* Wlo = Whi + (size_t)NDIM * KDIM;

  fourier_trig_kernel<<<(D1V * NMODES + 255) / 256, 256, 0, stream>>>(E, c, U, V);

  dim3 gw(KDIM / 256, NDIM / 32);
  fourier_weight_kernel<<<gw, 256, 0, stream>>>(Wb, U, V, Whi, Wlo);

  dim3 gg(NDIM / 128, MTOT / 128);
  fourier_gemm_kernel<<<gg, 256, 0, stream>>>(x, Whi, Wlo, bb, out);
}
